// DATMixFFNTransformerBlock_58325655879709
// MI455X (gfx1250) — matrix operand order repaired and checked by reading, not compiled, not run on hardware
//
#include <hip/hip_runtime.h>

// ---------------------------------------------------------------------------
// Types for CDNA5 (gfx1250) WMMA, wave32.
// ---------------------------------------------------------------------------
typedef __attribute__((ext_vector_type(16))) __bf16 v16bf;
typedef __attribute__((ext_vector_type(8)))  float  v8f;
typedef __attribute__((ext_vector_type(4)))  unsigned int u32x4;
typedef __attribute__((ext_vector_type(4)))  int i32x4;

#define AS1 __attribute__((address_space(1)))
#define AS3 __attribute__((address_space(3)))

// Async global->LDS path (CDNA5): guarded so the file still compiles if the
// toolchain lacks the builtins (falls back to VGPR staging).
#if defined(__AMDGCN__) && __has_builtin(__builtin_amdgcn_global_load_async_to_lds_b128)
#define USE_ASYNC_LDS 1
#else
#define USE_ASYNC_LDS 0
#endif

#if USE_ASYNC_LDS
// Builtin prototype (from hipcc diagnostics): (V4i addrspace(1)*, V4i
// addrspace(3)*, imm int offset, imm int cpol).  Generic LDS pointers carry
// the LDS byte offset in their low 32 bits, so cast via integers.
static __device__ __forceinline__ AS3 i32x4* lds_v4(void* p) {
  return (AS3 i32x4*)(unsigned int)(unsigned long long)p;
}
static __device__ __forceinline__ AS1 i32x4* glb_v4(const void* p) {
  return (AS1 i32x4*)(unsigned long long)p;
}
#endif

static __device__ __forceinline__ void wait_async0() {
#if USE_ASYNC_LDS
#if __has_builtin(__builtin_amdgcn_s_wait_asynccnt)
  __builtin_amdgcn_s_wait_asynccnt(0);
#else
  asm volatile("s_wait_asynccnt 0x0" ::: "memory");
#endif
#endif
}

static __device__ __forceinline__ __bf16 f2bf(float v) { return (__bf16)v; }
static __device__ __forceinline__ float  bf2f(__bf16 v) { return (float)v; }

static __device__ __forceinline__ float gelu_exact(float v) {
  return 0.5f * v * (1.0f + erff(v * 0.70710678118654752f));
}

// Fragment layouts used here (contiguous k, same in A and B; not the ISA table order):
//  A 16x32 bf16: lane L: m=L&15, h=L>>4; per lane = one contiguous 32B chunk
//    of the 64B k-row at byte offset h*32 (k=h*16+i).
//  B 32x16 bf16: lane L: n=L&15, h=L>>4; elements i=0..15 <-> k=h*16+i, i.e.
//    one contiguous 32B chunk of a k-major row (B staged transposed in LDS).
//  C/D f32: lane L: n=L&15, VGPR j: m=j+8*(L>>4).

// ---------------------------------------------------------------------------
// Channel LayerNorm over C at each (b,n); optional exact-GELU; optional
// "scatter" reproducing the reference's raw reshape(B,N,C) reinterpretation.
// ---------------------------------------------------------------------------
__launch_bounds__(256)
__global__ void k_ln_channel(const float* __restrict__ x,
                             const float* __restrict__ g,
                             const float* __restrict__ bta,
                             __bf16* __restrict__ outB,
                             float* __restrict__ outF,
                             int C, int N, int total, int do_gelu, int scatter) {
  int wave = threadIdx.x >> 5, lane = threadIdx.x & 31;
  int pos = blockIdx.x * 8 + wave;
  if (pos >= total) return;
  int b = pos / N, n = pos % N;
  const float* base = x + (size_t)b * C * N + n;
  float s = 0.f, s2 = 0.f;
  for (int c = lane; c < C; c += 32) {
    float v = base[(size_t)c * N];
    s += v; s2 += v * v;
  }
  for (int m = 1; m < 32; m <<= 1) {
    s  += __shfl_xor(s,  m, 32);
    s2 += __shfl_xor(s2, m, 32);
  }
  float mu = s / (float)C;
  float var = s2 / (float)C - mu * mu;
  float rstd = rsqrtf(var + 1e-5f);
  for (int c = lane; c < C; c += 32) {
    float v = (base[(size_t)c * N] - mu) * rstd * g[c] + bta[c];
    if (do_gelu) v = gelu_exact(v);
    size_t dst;
    if (scatter) {                       // flat j=c*1024+n -> (j%768)*1024 + j/768
      size_t j = (size_t)c * 1024 + n;
      dst = (size_t)b * C * N + (j % 768) * 1024 + (j / 768);
    } else {
      dst = ((size_t)b * C + c) * N + n;
    }
    if (outB) outB[dst] = f2bf(v);
    if (outF) outF[dst] = v;
  }
}

// ---------------------------------------------------------------------------
// fp32 -> bf16 weight conversion.
// ---------------------------------------------------------------------------
__launch_bounds__(256)
__global__ void k_cvt(const float* __restrict__ in, __bf16* __restrict__ out, int n) {
  int i = blockIdx.x * 256 + threadIdx.x;
  if (i < n) out[i] = f2bf(in[i]);
}

// ---------------------------------------------------------------------------
// Batched GEMM: out[b][m][n] = sum_k A[m][k]*Bact[b][k][n] + bias[m] (+resid).
// bf16 WMMA, f32 accumulation.  Block tile 128(M)x128(N), 8 waves each owning
// 32x64 (2x4 wmma tiles, 8 v_wmma per K step).  K stepped by 32 through
// double-buffered LDS; the A tile is staged with GLOBAL_LOAD_ASYNC_TO_LDS
// (ASYNCcnt) overlapping the WMMA work on the current buffer.
// swz=1 writes the scrambled (n*M+m) layout used by fc2.
// ---------------------------------------------------------------------------
__launch_bounds__(256)
__global__ void k_gemm(const __bf16* __restrict__ A,
                       const __bf16* __restrict__ Bact,
                       const float* __restrict__ bias,
                       const float* __restrict__ resid,
                       float* __restrict__ outF,
                       __bf16* __restrict__ outB,
                       int M, int K, int N, int swz) {
  __shared__ __align__(16) __bf16 lA[2][128 * 32];    // [m][k], 64B rows
  __shared__ __align__(16) __bf16 lBT[2][128 * 32];   // [n][k] transposed, 64B rows

  const int t = threadIdx.x;
  const int wave = t >> 5, lane = t & 31;
  const int h = lane >> 4, l16 = lane & 15;
  const int batch = blockIdx.z;
  const int m0 = blockIdx.y * 128;
  const int n0 = blockIdx.x * 128;
  const int wm = (wave >> 1) * 32;    // 4 waves along M
  const int wn = (wave & 1) * 64;     // 2 waves along N

  const __bf16* Bb = Bact + (size_t)batch * K * N;

  const int arow = t >> 1, ahk = t & 1;          // A stage: 32B per thread
  const int bkk = t >> 3, bnb = (t & 7) * 16;    // B stage: 32B per thread

  const __bf16* aSrc = A + (size_t)(m0 + arow) * K + ahk * 16;
  const __bf16* bSrc = Bb + (size_t)bkk * N + n0 + bnb;

  v8f acc[2][4] = {};

  auto stageA = [&](int k0, int buf) {
    const __bf16* g = aSrc + k0;
    __bf16* l = &lA[buf][arow * 32 + ahk * 16];
#if USE_ASYNC_LDS
    __builtin_amdgcn_global_load_async_to_lds_b128(glb_v4(g), lds_v4(l), 0, 0);
    __builtin_amdgcn_global_load_async_to_lds_b128(glb_v4(g + 8), lds_v4(l + 8), 0, 0);
#else
    const u32x4* src = (const u32x4*)g;
    u32x4 d0 = src[0], d1 = src[1];
    u32x4* dst = (u32x4*)l;
    dst[0] = d0; dst[1] = d1;
#endif
    __builtin_prefetch(g + 64, 0, 1);           // next-next K slab (speculative)
  };

  auto stageB = [&](int k0, int buf) {
    const __bf16* src = bSrc + (size_t)k0 * N;
    union { u32x4 u; __bf16 e[8]; } u0, u1;
    u0.u = ((const u32x4*)src)[0];
    u1.u = ((const u32x4*)src)[1];
    __builtin_prefetch(src + (size_t)32 * N, 0, 1);
    __bf16* dst = &lBT[buf][0];
#pragma unroll
    for (int e = 0; e < 8; ++e) dst[(bnb + e) * 32 + bkk] = u0.e[e];
#pragma unroll
    for (int e = 0; e < 8; ++e) dst[(bnb + 8 + e) * 32 + bkk] = u1.e[e];
  };

  const int nk = K >> 5;
  stageA(0, 0);
  stageB(0, 0);
  wait_async0();
  __syncthreads();

  for (int i = 0; i < nk; ++i) {
    const int cur = i & 1;
    if (i + 1 < nk) {                 // stage next tile into the other buffer
      stageA((i + 1) << 5, cur ^ 1);
      stageB((i + 1) << 5, cur ^ 1);
    }
    v16bf a[2], bf4[4];
#pragma unroll
    for (int mi = 0; mi < 2; ++mi) {
      const u32x4* p = (const u32x4*)&lA[cur][(wm + mi * 16 + l16) * 32];
      union { u32x4 u[2]; v16bf v; } un;
      un.u[0] = p[h * 2]; un.u[1] = p[h * 2 + 1];
      a[mi] = un.v;
    }
#pragma unroll
    for (int ni = 0; ni < 4; ++ni) {
      const u32x4* p = (const u32x4*)&lBT[cur][(wn + ni * 16 + l16) * 32];
      union { u32x4 u[2]; v16bf v; } un;
      un.u[0] = p[h * 2]; un.u[1] = p[h * 2 + 1];
      bf4[ni] = un.v;
    }
#pragma unroll
    for (int mi = 0; mi < 2; ++mi)
#pragma unroll
      for (int ni = 0; ni < 4; ++ni)
        acc[mi][ni] = __builtin_amdgcn_wmma_f32_16x16x32_bf16(
            false, a[mi], false, bf4[ni], (short)0, acc[mi][ni], false, false);
    wait_async0();
    __syncthreads();
  }

  // Epilogue.
#pragma unroll
  for (int mi = 0; mi < 2; ++mi)
#pragma unroll
    for (int ni = 0; ni < 4; ++ni) {
      int gn = n0 + wn + ni * 16 + l16;
#pragma unroll
      for (int j = 0; j < 8; ++j) {
        int gm = m0 + wm + mi * 16 + j + 8 * h;
        float v = acc[mi][ni][j] + bias[gm];
        size_t idx = swz ? ((size_t)batch * M * N + (size_t)gn * M + gm)
                         : (((size_t)batch * M + gm) * N + gn);
        if (resid) v += resid[idx];
        if (outF) outF[idx] = v;
        if (outB) outB[idx] = f2bf(v);
      }
    }
}

// ---------------------------------------------------------------------------
// Transpose [96][32][1024] f32 -> [96][1024][32] bf16 (Q^T / K^T fragments).
// ---------------------------------------------------------------------------
__launch_bounds__(256)
__global__ void k_transpose32(const float* __restrict__ in, __bf16* __restrict__ out) {
  int g = blockIdx.x;
  int n = blockIdx.y * 256 + threadIdx.x;
  const float* src = in + (size_t)g * 32 * 1024 + n;
  __bf16* dst = out + ((size_t)g * 1024 + n) * 32;
#pragma unroll
  for (int c = 0; c < 32; ++c) dst[c] = f2bf(src[(size_t)c * 1024]);
}

// ---------------------------------------------------------------------------
// Deformable offset branch + bilinear sampling.  One block per (b*G+g, row h):
// 9x9 depthwise conv over q (24,128,32,32) -> channel LN -> GELU -> 2x128
// pointwise -> tanh/scale -> grid -> bilinear sample of LN(x) -> x_samp bf16.
// ---------------------------------------------------------------------------
__launch_bounds__(256)
__global__ void k_offset_sample(const float* __restrict__ q,
                                const __bf16* __restrict__ xn,
                                const float* __restrict__ dww,
                                const float* __restrict__ dwb,
                                const float* __restrict__ lng,
                                const float* __restrict__ lnb,
                                const float* __restrict__ pww,
                                __bf16* __restrict__ xsamp) {
  __shared__ float o[128 * 32];
  __shared__ float reds[256], reds2[256];
  __shared__ float mus[32], rstds[32], gxs[32], gys[32];

  const int t = threadIdx.x;
  const int bg = blockIdx.x;         // 0..23
  const int hh = blockIdx.y;         // 0..31
  const int b = bg / 6, g = bg % 6;
  const float* qbase = q + (size_t)(b * 768 + g * 128) * 1024;

  // 9x9 depthwise conv (pad 4).
#pragma unroll 1
  for (int it = 0; it < 16; ++it) {
    int idx = t + it * 256;
    int cg = idx >> 5, w = idx & 31;
    float acc = dwb[cg];
    const float* qc = qbase + (size_t)cg * 1024;
#pragma unroll 1
    for (int dy = 0; dy < 9; ++dy) {
      int yy = hh + dy - 4;
      if (yy < 0 || yy > 31) continue;
#pragma unroll 1
      for (int dx = 0; dx < 9; ++dx) {
        int xx = w + dx - 4;
        if (xx < 0 || xx > 31) continue;
        acc += qc[yy * 32 + xx] * dww[cg * 81 + dy * 9 + dx];
      }
    }
    o[cg * 32 + w] = acc;
  }
  __syncthreads();

  // Channel LN over cg (128) per column w.
  {
    int w = t >> 3, r = t & 7;
    float s = 0.f, s2 = 0.f;
    for (int cg = r * 16; cg < r * 16 + 16; ++cg) {
      float v = o[cg * 32 + w]; s += v; s2 += v * v;
    }
    reds[w * 8 + r] = s; reds2[w * 8 + r] = s2;
  }
  __syncthreads();
  if (t < 32) {
    float s = 0.f, s2 = 0.f;
    for (int r = 0; r < 8; ++r) { s += reds[t * 8 + r]; s2 += reds2[t * 8 + r]; }
    float mu = s / 128.f;
    mus[t] = mu;
    rstds[t] = rsqrtf(s2 / 128.f - mu * mu + 1e-5f);
  }
  __syncthreads();
#pragma unroll 1
  for (int it = 0; it < 16; ++it) {
    int idx = t + it * 256;
    int cg = idx >> 5, w = idx & 31;
    float v = (o[cg * 32 + w] - mus[w]) * rstds[w] * lng[cg] + lnb[cg];
    o[cg * 32 + w] = gelu_exact(v);
  }
  __syncthreads();

  // 2x128 pointwise -> tanh * 1/31 -> absolute sample coordinates.
  if (t < 64) {
    int oo = t >> 5, w = t & 31;
    float s = 0.f;
    for (int cg = 0; cg < 128; ++cg) s += pww[oo * 128 + cg] * o[cg * 32 + w];
    float off = tanhf(s) * (1.0f / 31.0f);
    if (oo == 0) {                                   // channel 0 = y
      float ry = ((float)hh + 0.5f) / 32.f * 2.f - 1.f;
      gys[w] = (off + ry + 1.f) * 0.5f * 31.f;
    } else {                                         // channel 1 = x
      float rx = ((float)w + 0.5f) / 32.f * 2.f - 1.f;
      gxs[w] = (off + rx + 1.f) * 0.5f * 31.f;
    }
  }
  __syncthreads();

  // Bilinear sample of LN(x) (group bg) with zero-padding OOB.
  const __bf16* xb = xn + (size_t)(b * 768 + g * 128) * 1024;
#pragma unroll 1
  for (int it = 0; it < 16; ++it) {
    int idx = t + it * 256;
    int cg = idx >> 5, w = idx & 31;
    float fx = gxs[w], fy = gys[w];
    float x0f = floorf(fx), y0f = floorf(fy);
    int x0 = (int)x0f, y0 = (int)y0f, x1 = x0 + 1, y1 = y0 + 1;
    float wx1 = fx - x0f, wx0 = 1.f - wx1;
    float wy1 = fy - y0f, wy0 = 1.f - wy1;
    const __bf16* ch = xb + (size_t)cg * 1024;
    float v00 = (x0 >= 0 && x0 <= 31 && y0 >= 0 && y0 <= 31) ? bf2f(ch[y0 * 32 + x0]) : 0.f;
    float v10 = (x1 >= 0 && x1 <= 31 && y0 >= 0 && y0 <= 31) ? bf2f(ch[y0 * 32 + x1]) : 0.f;
    float v01 = (x0 >= 0 && x0 <= 31 && y1 >= 0 && y1 <= 31) ? bf2f(ch[y1 * 32 + x0]) : 0.f;
    float v11 = (x1 >= 0 && x1 <= 31 && y1 >= 0 && y1 <= 31) ? bf2f(ch[y1 * 32 + x1]) : 0.f;
    float val = v00 * wx0 * wy0 + v10 * wx1 * wy0 + v01 * wx0 * wy1 + v11 * wx1 * wy1;
    xsamp[((size_t)(b * 768 + g * 128) + cg) * 1024 + hh * 32 + w] = f2bf(val);
  }
}

// ---------------------------------------------------------------------------
// Attention, flash-style online softmax.  HC=32 == one WMMA K step.
// qT/kT: [96][1024][32] bf16 (fragment-contiguous).  v: [b][768][1024] bf16.
// Block = 8 waves, each owns 16 query rows; grid (8, 96).
// ---------------------------------------------------------------------------
__launch_bounds__(256)
__global__ void k_attn(const __bf16* __restrict__ qT,
                       const __bf16* __restrict__ kT,
                       const __bf16* __restrict__ v,
                       __bf16* __restrict__ out) {
  __shared__ __align__(16) __bf16 lP[8 * 16 * 32];   // per-wave 16x32 P staging

  const int t = threadIdx.x, wave = t >> 5, lane = t & 31;
  const int h = lane >> 4, l16 = lane & 15;
  const int gh = blockIdx.y;               // b*24 + head
  const int b = gh / 24, head = gh % 24;
  const int m0 = blockIdx.x * 128 + wave * 16;
  const float scale = 0.17677669529663687f;   // 32^-0.5

  // Q A-fragment (covers all 32 channels; loaded once).
  v16bf aq;
  {
    const u32x4* p = (const u32x4*)(qT + ((size_t)gh * 1024 + m0 + l16) * 32);
    union { u32x4 u[2]; v16bf v; } un;
    un.u[0] = p[h * 2]; un.u[1] = p[h * 2 + 1];
    aq = un.v;
  }

  v8f o0 = {}, o1 = {};
  float rm[8], rs[8];
#pragma unroll
  for (int j = 0; j < 8; ++j) { rm[j] = -3.0e38f; rs[j] = 0.f; }

  __bf16* Pw = lP + wave * 512;
  const __bf16* kbase = kT + (size_t)gh * 1024 * 32;
  const __bf16* vbase = v + (size_t)(b * 768 + head * 32) * 1024;

  for (int nc = 0; nc < 1024; nc += 32) {
    // K B-fragments for two 16-wide score tiles.
    v16bf bk0, bk1;
    {
      const u32x4* p = (const u32x4*)(kbase + (size_t)(nc + l16) * 32);
      union { u32x4 u[2]; v16bf v; } un;
      un.u[0] = p[h * 2]; un.u[1] = p[h * 2 + 1];
      bk0 = un.v;
      const u32x4* p2 = (const u32x4*)(kbase + (size_t)(nc + 16 + l16) * 32);
      un.u[0] = p2[h * 2]; un.u[1] = p2[h * 2 + 1];
      bk1 = un.v;
    }
    v8f zero = {};
    v8f s0 = __builtin_amdgcn_wmma_f32_16x16x32_bf16(false, aq, false, bk0, (short)0, zero, false, false);
    v8f s1 = __builtin_amdgcn_wmma_f32_16x16x32_bf16(false, aq, false, bk1, (short)0, zero, false, false);

    // Online softmax update + stage P (bf16) into LDS in A-fragment-ready rows.
#pragma unroll
    for (int j = 0; j < 8; ++j) {
      float x0 = s0[j] * scale, x1 = s1[j] * scale;
      float mx = fmaxf(x0, x1);
      mx = fmaxf(mx, __shfl_xor(mx, 1, 32));
      mx = fmaxf(mx, __shfl_xor(mx, 2, 32));
      mx = fmaxf(mx, __shfl_xor(mx, 4, 32));
      mx = fmaxf(mx, __shfl_xor(mx, 8, 32));
      float mnew = fmaxf(rm[j], mx);
      float corr = __expf(rm[j] - mnew);
      rm[j] = mnew;
      float p0 = __expf(x0 - mnew), p1 = __expf(x1 - mnew);
      float ps = p0 + p1;
      ps += __shfl_xor(ps, 1, 32);
      ps += __shfl_xor(ps, 2, 32);
      ps += __shfl_xor(ps, 4, 32);
      ps += __shfl_xor(ps, 8, 32);
      rs[j] = rs[j] * corr + ps;
      o0[j] *= corr; o1[j] *= corr;
      int m = j + 8 * h;
      Pw[m * 32 + l16] = f2bf(p0);
      Pw[m * 32 + 16 + l16] = f2bf(p1);
    }
    asm volatile("" ::: "memory");   // keep DS store->load order (same-wave DS in-order)

    // P A-fragment back from LDS.
    v16bf ap;
    {
      const u32x4* p = (const u32x4*)(Pw + l16 * 32);
      union { u32x4 u[2]; v16bf v; } un;
      un.u[0] = p[h * 2]; un.u[1] = p[h * 2 + 1];
      ap = un.v;
    }
    // V B-fragments (k-dim = n chunk, contiguous along n in v[c][n]).
    v16bf bv0, bv1;
    {
      const u32x4* p = (const u32x4*)(vbase + (size_t)l16 * 1024 + nc + h * 16);
      union { u32x4 u[2]; v16bf v; } un;
      un.u[0] = p[0]; un.u[1] = p[1];
      bv0 = un.v;
      const u32x4* p2 = (const u32x4*)(vbase + (size_t)(16 + l16) * 1024 + nc + h * 16);
      un.u[0] = p2[0]; un.u[1] = p2[1];
      bv1 = un.v;
    }
    o0 = __builtin_amdgcn_wmma_f32_16x16x32_bf16(false, ap, false, bv0, (short)0, o0, false, false);
    o1 = __builtin_amdgcn_wmma_f32_16x16x32_bf16(false, ap, false, bv1, (short)0, o1, false, false);
  }

  // Normalize and store bf16 out[b][head*32+c][m].
#pragma unroll
  for (int tile = 0; tile < 2; ++tile) {
    int c = head * 32 + tile * 16 + l16;
#pragma unroll
    for (int j = 0; j < 8; ++j) {
      int m = m0 + j + 8 * h;
      float val = (tile ? o1[j] : o0[j]) / rs[j];
      out[((size_t)b * 768 + c) * 1024 + m] = f2bf(val);
    }
  }
}

// ---------------------------------------------------------------------------
// 3x3 depthwise conv over h1 (B,3072,32,32) + bias + residual h1 -> ax.
// ---------------------------------------------------------------------------
__launch_bounds__(256)
__global__ void k_dwconv3x3_add(const float* __restrict__ h1,
                                const float* __restrict__ w,
                                const float* __restrict__ bias,
                                float* __restrict__ ax) {
  size_t i = (size_t)blockIdx.x * 256 + threadIdx.x;   // over 4*3072*1024
  int n = (int)(i & 1023);
  size_t cb = i >> 10;
  int c = (int)(cb % 3072);
  int y = n >> 5, x = n & 31;
  const float* base = h1 + (cb << 10);
  float acc = bias[c];
#pragma unroll
  for (int dy = -1; dy <= 1; ++dy) {
    int yy = y + dy;
    if (yy < 0 || yy > 31) continue;
#pragma unroll
    for (int dx = -1; dx <= 1; ++dx) {
      int xx = x + dx;
      if (xx < 0 || xx > 31) continue;
      acc += base[yy * 32 + xx] * w[c * 9 + (dy + 1) * 3 + (dx + 1)];
    }
  }
  ax[i] = acc + h1[i];
}

// ---------------------------------------------------------------------------
// Host-side orchestration.
// ---------------------------------------------------------------------------
extern "C" void kernel_launch(void* const* d_in, const int* in_sizes, int n_in,
                              void* d_out, int out_size, void* d_ws, size_t ws_size,
                              hipStream_t stream) {
  const float* x       = (const float*)d_in[0];
  const float* ln1_g   = (const float*)d_in[1];
  const float* ln1_b   = (const float*)d_in[2];
  const float* off_dw_w= (const float*)d_in[3];
  const float* off_dw_b= (const float*)d_in[4];
  const float* off_ln_g= (const float*)d_in[5];
  const float* off_ln_b= (const float*)d_in[6];
  const float* off_pw_w= (const float*)d_in[7];
  const float* wq = (const float*)d_in[8];   const float* bq = (const float*)d_in[9];
  const float* wk = (const float*)d_in[10];  const float* bk = (const float*)d_in[11];
  const float* wv = (const float*)d_in[12];  const float* bv = (const float*)d_in[13];
  const float* wo = (const float*)d_in[14];  const float* bo = (const float*)d_in[15];
  const float* ln2_g = (const float*)d_in[16];
  const float* ln2_b = (const float*)d_in[17];
  const float* fc1_w = (const float*)d_in[18];
  const float* fc1_b = (const float*)d_in[19];
  const float* dw_w  = (const float*)d_in[20];
  const float* dw_b  = (const float*)d_in[21];
  const float* ffn_ln_g = (const float*)d_in[22];
  const float* ffn_ln_b = (const float*)d_in[23];
  const float* fc2_w = (const float*)d_in[24];
  const float* fc2_b = (const float*)d_in[25];
  float* out = (float*)d_out;

  char* wsp = (char*)d_ws;
  size_t off = 0;
  auto carve = [&](size_t bytes) -> void* {
    void* p = wsp + off;
    off += (bytes + 255) & ~(size_t)255;
    return p;
  };
  const size_t AN = (size_t)4 * 768 * 1024;        // activation elements
  __bf16* wq_bf  = (__bf16*)carve(768 * 768 * 2);
  __bf16* wk_bf  = (__bf16*)carve(768 * 768 * 2);
  __bf16* wv_bf  = (__bf16*)carve(768 * 768 * 2);
  __bf16* wo_bf  = (__bf16*)carve(768 * 768 * 2);
  __bf16* fc1_bf = (__bf16*)carve((size_t)3072 * 768 * 2);
  __bf16* fc2_bf = (__bf16*)carve((size_t)3072 * 768 * 2);
  __bf16* xn_bf  = (__bf16*)carve(AN * 2);
  float*  q_f    = (float*)carve(AN * 4);
  __bf16* xs_bf  = (__bf16*)carve(AN * 2);
  __bf16* qT_bf  = (__bf16*)carve(AN * 2);
  float*  k_f    = (float*)carve(AN * 4);
  __bf16* kT_bf  = (__bf16*)carve(AN * 2);
  __bf16* v_bf   = (__bf16*)carve(AN * 2);
  __bf16* at_bf  = (__bf16*)carve(AN * 2);
  float*  x1_f   = (float*)carve(AN * 4);
  __bf16* t_bf   = (__bf16*)carve(AN * 2);
  float*  h1_f   = (float*)carve((size_t)4 * 3072 * 1024 * 4);
  float*  ax_f   = (float*)carve((size_t)4 * 3072 * 1024 * 4);
  __bf16* g_bf   = (__bf16*)carve((size_t)4 * 3072 * 1024 * 2);

  // Weight conversions fp32 -> bf16.
  k_cvt<<<(768 * 768 + 255) / 256, 256, 0, stream>>>(wq, wq_bf, 768 * 768);
  k_cvt<<<(768 * 768 + 255) / 256, 256, 0, stream>>>(wk, wk_bf, 768 * 768);
  k_cvt<<<(768 * 768 + 255) / 256, 256, 0, stream>>>(wv, wv_bf, 768 * 768);
  k_cvt<<<(768 * 768 + 255) / 256, 256, 0, stream>>>(wo, wo_bf, 768 * 768);
  k_cvt<<<(3072 * 768 + 255) / 256, 256, 0, stream>>>(fc1_w, fc1_bf, 3072 * 768);
  k_cvt<<<(3072 * 768 + 255) / 256, 256, 0, stream>>>(fc2_w, fc2_bf, 3072 * 768);

  // LN1: x -> xn (bf16).
  k_ln_channel<<<512, 256, 0, stream>>>(x, ln1_g, ln1_b, xn_bf, nullptr,
                                        768, 1024, 4096, 0, 0);
  // q = wq @ xn + bq  (fp32, feeds offset branch + qT).
  k_gemm<<<dim3(8, 6, 4), 256, 0, stream>>>(wq_bf, xn_bf, bq, nullptr,
                                            q_f, nullptr, 768, 768, 1024, 0);
  // Deformable offsets + sampling -> x_samp (bf16).
  k_offset_sample<<<dim3(24, 32), 256, 0, stream>>>(q_f, xn_bf, off_dw_w, off_dw_b,
                                                    off_ln_g, off_ln_b, off_pw_w, xs_bf);
  // qT for attention fragments.
  k_transpose32<<<dim3(96, 4), 256, 0, stream>>>(q_f, qT_bf);
  // k = wk @ x_samp + bk -> kT.
  k_gemm<<<dim3(8, 6, 4), 256, 0, stream>>>(wk_bf, xs_bf, bk, nullptr,
                                            k_f, nullptr, 768, 768, 1024, 0);
  k_transpose32<<<dim3(96, 4), 256, 0, stream>>>(k_f, kT_bf);
  // v = wv @ x_samp + bv (bf16 only).
  k_gemm<<<dim3(8, 6, 4), 256, 0, stream>>>(wv_bf, xs_bf, bv, nullptr,
                                            nullptr, v_bf, 768, 768, 1024, 0);
  // Attention.
  k_attn<<<dim3(8, 96), 256, 0, stream>>>(qT_bf, kT_bf, v_bf, at_bf);
  // x1 = wo @ attn_out + bo + x.
  k_gemm<<<dim3(8, 6, 4), 256, 0, stream>>>(wo_bf, at_bf, bo, x,
                                            x1_f, nullptr, 768, 768, 1024, 0);
  // LN2 with the reference's raw reshape(B,N,C) reinterpretation (scatter).
  k_ln_channel<<<512, 256, 0, stream>>>(x1_f, ln2_g, ln2_b, t_bf, nullptr,
                                        768, 1024, 4096, 0, 1);
  // h1 = fc1 @ t + fc1_b  (output layout already = hc2 (B,3072,32,32)).
  k_gemm<<<dim3(8, 24, 4), 256, 0, stream>>>(fc1_bf, t_bf, fc1_b, nullptr,
                                             h1_f, nullptr, 3072, 768, 1024, 0);
  // ax = dwconv3x3(h1) + dw_b + h1.
  k_dwconv3x3_add<<<49152, 256, 0, stream>>>(h1_f, dw_w, dw_b, ax_f);
  // FFN LayerNorm over DFF + exact GELU -> bf16.
  k_ln_channel<<<512, 256, 0, stream>>>(ax_f, ffn_ln_g, ffn_ln_b, g_bf, nullptr,
                                        3072, 1024, 4096, 1, 0);
  // o2 = fc2 @ g + fc2_b, written through the reference's reshape scramble,
  // plus residual x1 -> final output.
  k_gemm<<<dim3(8, 6, 4), 256, 0, stream>>>(fc2_bf, g_bf, fc2_b, x1_f,
                                            out, nullptr, 768, 3072, 1024, 1);
}
